// WeLMMoeDecoderLayer_31576599560864
// MI455X (gfx1250) — matrix operand order repaired and checked by reading, not compiled, not run on hardware
//
#include <hip/hip_runtime.h>

// ---------------------------------------------------------------------------
// MI455X (gfx1250) implementation of the WeLM MoE decoder layer.
// All dense GEMMs run on v_wmma_f32_16x16x32_bf16 (wave32 WMMA), attention is
// a flash-attention kernel whose QK^T and P*V products are also WMMA.
// Weights stay fp32 in HBM (memory-bound: ~460MB/pass @ 23.3TB/s) and are
// converted to bf16 while staged into LDS; activations are carried as bf16.
// k_gemm double-buffers the weight tile in VGPRs (global loads for tile k+1
// issue before the WMMAs of tile k), commits it with packed ds_store_b128,
// and DMA's the bf16 A tile into LDS with global_load_async_to_lds_b128 /
// s_wait_asynccnt (CDNA5 async path).
// ---------------------------------------------------------------------------

typedef __attribute__((ext_vector_type(16))) __bf16 v16bf;
typedef __attribute__((ext_vector_type(8)))  float  v8f;

static __device__ inline unsigned short f2bf(float x) {
  unsigned int u = __float_as_uint(x);
  u += 0x7FFFu + ((u >> 16) & 1u);          // round-to-nearest-even
  return (unsigned short)(u >> 16);
}
static __device__ inline unsigned pk2bf(float lo, float hi) {
  return (unsigned)f2bf(lo) | ((unsigned)f2bf(hi) << 16);
}
static __device__ inline float bf2f(unsigned short b) {
  return __uint_as_float(((unsigned int)b) << 16);
}
static __device__ inline v8f zero8() {
  v8f z;
#pragma unroll
  for (int i = 0; i < 8; i++) z[i] = 0.f;
  return z;
}

union FragU {
  v16bf v;
  uint4 q[2];
};

// A-matrix fragment: 16x32 bf16 tile, row-major source with `stride` elements.
// Per ISA 7.12.2: lane<16 holds row=lane, K = {0..7,16..23}; lane>=16 holds
// row=lane-16, K = {8..15,24..31}.  Two 128-bit loads per lane.
static __device__ inline v16bf load_frag_a(const unsigned short* p, int stride) {
  int lane = threadIdx.x & 31;
  int half = lane >> 4;
  int row  = lane & 15;
  const unsigned short* s0 = p + (size_t)row * stride + half * 8;
  FragU f;
  f.q[0] = *(const uint4*)(s0);
  f.q[1] = *(const uint4*)(s0 + 16);
  return f.v;
}

// B-matrix fragment: 32x16 bf16 tile, provided TRANSPOSED (BT[n][k], stride
// elems).  lane<16 -> col n=lane, K={0..7,16..23}; lane>=16 -> col n=lane-16, K={8..15,24..31}.
static __device__ inline v16bf load_frag_b(const unsigned short* p, int stride) {
  int lane = threadIdx.x & 31;
  int half = lane >> 4;
  int col  = lane & 15;
  const unsigned short* s0 = p + (size_t)col * stride + half * 8;
  FragU f;
  f.q[0] = *(const uint4*)(s0);
  f.q[1] = *(const uint4*)(s0 + 16);
  return f.v;
}

static __device__ inline v8f wmma_bf16(v16bf a, v16bf b, v8f c) {
  return __builtin_amdgcn_wmma_f32_16x16x32_bf16(false, a, false, b,
                                                 (short)0, c, false, false);
}

// ---------------------------------------------------------------------------
// RMSNorm: f32 in -> bf16 out (x * rsqrt(mean(x^2)+eps) * w)
// ---------------------------------------------------------------------------
__global__ __launch_bounds__(256) void k_rms(const float* __restrict__ x,
                                             const float* __restrict__ w,
                                             unsigned short* __restrict__ out,
                                             int H) {
  int t = blockIdx.x;
  const float* xr = x + (size_t)t * H;
  __shared__ float red[256];
  float s = 0.f;
  for (int i = threadIdx.x; i < H; i += 256) { float v = xr[i]; s += v * v; }
  red[threadIdx.x] = s;
  __syncthreads();
  for (int o = 128; o > 0; o >>= 1) {
    if (threadIdx.x < o) red[threadIdx.x] += red[threadIdx.x + o];
    __syncthreads();
  }
  float inv = rsqrtf(red[0] / (float)H + 1e-6f);
  // packed bf16 pair stores (H is even)
  for (int i = threadIdx.x * 2; i < H; i += 512) {
    unsigned p = pk2bf(xr[i] * inv * w[i], xr[i + 1] * inv * w[i + 1]);
    *(unsigned*)&out[(size_t)t * H + i] = p;
  }
}

// ---------------------------------------------------------------------------
// Generic WMMA GEMM: C[M,N] = A_bf16[M,K] * W_f32[K,N]  (+fused epilogues)
// 128x128 tile / 256 threads (8 waves, 4x2), BK=32 (one WMMA K-step).
//  - W tile: fp32 global -> VGPR (double-buffered; next tile's loads issue
//    before current tile's WMMAs) -> f2bf pack -> LDS transposed (stride 40,
//    two ds_store_b128 per thread) so B fragments are two ds_load_b128.
//  - A tile: bf16 global -> LDS via global_load_async_to_lds_b128 (ASYNCcnt).
// mode 0: store     1: +bias[col]      2: +aux[row*N+col] (residual)
// mode 3: *sigmoid(aux[row])           4: C += v * aux[row*aux_stride+aux_off]
// ---------------------------------------------------------------------------
__global__ __launch_bounds__(256) void k_gemm(
    const unsigned short* __restrict__ A, const float* __restrict__ W,
    float* __restrict__ C, int M, int N, int K, int mode,
    const float* __restrict__ bias, const float* __restrict__ aux,
    int aux_stride, int aux_off) {
  const int n0 = blockIdx.x * 128;
  const int m0 = blockIdx.y * 128;
  const int tid = threadIdx.x;
  const int wave = tid >> 5, lane = tid & 31;
  const int wm = wave >> 1, wn = wave & 1;
  const int half = lane >> 4, l15 = lane & 15;

  __shared__ unsigned short Wsh[128 * 40];  // [n][k], padded stride 40
  __shared__ unsigned short Ash[128 * 32];  // [m][k] bf16 A tile, stride 32

  // W staging ownership: thread owns row nn, 16 contiguous k-elements.
  const int nn = tid & 127;
  const int kkb = (tid >> 7) * 16;

  // LDS byte address of this thread's async-copy destination (low 32 bits of
  // a flat pointer into the LDS aperture are the wave-relative LDS address).
  // Thread t copies 32 contiguous bytes: row = t>>1, halfrow = t&1.
  const unsigned ash_dst =
      (unsigned)(size_t)(&Ash[0]) + (unsigned)(tid * 32);

  v8f acc[2][4];
#pragma unroll
  for (int mt = 0; mt < 2; mt++)
#pragma unroll
    for (int nt = 0; nt < 4; nt++) acc[mt][nt] = zero8();

  float wreg[16];
#pragma unroll
  for (int j = 0; j < 16; j++)
    wreg[j] = W[(size_t)(kkb + j) * N + n0 + nn];

  for (int k0 = 0; k0 < K; k0 += 32) {
    __syncthreads();  // previous iteration's LDS readers done
    // commit current W tile: pack 16 bf16 -> two ds_store_b128
    {
      uint4 q0, q1;
      q0.x = pk2bf(wreg[0], wreg[1]);
      q0.y = pk2bf(wreg[2], wreg[3]);
      q0.z = pk2bf(wreg[4], wreg[5]);
      q0.w = pk2bf(wreg[6], wreg[7]);
      q1.x = pk2bf(wreg[8], wreg[9]);
      q1.y = pk2bf(wreg[10], wreg[11]);
      q1.z = pk2bf(wreg[12], wreg[13]);
      q1.w = pk2bf(wreg[14], wreg[15]);
      uint4* dst = (uint4*)&Wsh[nn * 40 + kkb];
      dst[0] = q0;
      dst[1] = q1;
    }
    // async DMA of the bf16 A tile straight into LDS (no VGPR round-trip)
    {
      const unsigned short* asrc =
          A + (size_t)(m0 + (tid >> 1)) * K + k0 + (tid & 1) * 16;
      unsigned long long ga = (unsigned long long)(size_t)asrc;
      asm volatile(
          "global_load_async_to_lds_b128 %0, %1, off\n\t"
          "global_load_async_to_lds_b128 %0, %1, off offset:16"
          :: "v"(ash_dst), "v"(ga) : "memory");
    }
    // prefetch next W tile into registers; these global loads' latency is
    // hidden under the WMMA block below (consumed only at next commit)
    if (k0 + 32 < K) {
#pragma unroll
      for (int j = 0; j < 16; j++)
        wreg[j] = W[(size_t)(k0 + 32 + kkb + j) * N + n0 + nn];
    }
    asm volatile("s_wait_asynccnt 0x0" ::: "memory");
    __syncthreads();

    v16bf bfr[4];
#pragma unroll
    for (int nt = 0; nt < 4; nt++)
      bfr[nt] = load_frag_b(Wsh + (wn * 64 + nt * 16) * 40, 40);
#pragma unroll
    for (int mt = 0; mt < 2; mt++) {
      v16bf af = load_frag_a(Ash + (wm * 32 + mt * 16) * 32, 32);
#pragma unroll
      for (int nt = 0; nt < 4; nt++)
        acc[mt][nt] = wmma_bf16(af, bfr[nt], acc[mt][nt]);
    }
  }

  // epilogue (C layout: m = r + 8*half, n = l15)
#pragma unroll
  for (int mt = 0; mt < 2; mt++) {
#pragma unroll
    for (int nt = 0; nt < 4; nt++) {
#pragma unroll
      for (int r = 0; r < 8; r++) {
        int row = m0 + wm * 32 + mt * 16 + half * 8 + r;
        int col = n0 + wn * 64 + nt * 16 + l15;
        float v = acc[mt][nt][r];
        size_t o = (size_t)row * N + col;
        if (mode == 0)      C[o] = v;
        else if (mode == 1) C[o] = v + bias[col];
        else if (mode == 2) C[o] = v + aux[o];
        else if (mode == 3) C[o] = v * (1.f / (1.f + __expf(-aux[row])));
        else                C[o] += v * aux[(size_t)row * aux_stride + aux_off];
      }
    }
  }
}

// ---------------------------------------------------------------------------
// RoPE, in place on the f32 qkv buffer (16 Q heads + 4 K heads, half=64)
// ---------------------------------------------------------------------------
__global__ __launch_bounds__(256) void k_rope(float* __restrict__ qkv,
                                              const int* __restrict__ positions,
                                              int T) {
  int idx = blockIdx.x * 256 + threadIdx.x;
  if (idx >= T * 20 * 64) return;
  int i = idx & 63;
  int hh = (idx >> 6) % 20;
  int t = idx / (20 * 64);
  float invf = __powf(10000.0f, -(float)i * (1.0f / 64.0f));
  float f = (float)positions[t] * invf;
  float s, c;
  __sincosf(f, &s, &c);
  size_t base = (size_t)t * 3072 + (hh < 16 ? hh * 128 : 2048 + (hh - 16) * 128);
  float x1 = qkv[base + i], x2 = qkv[base + 64 + i];
  qkv[base + i]      = x1 * c - x2 * s;
  qkv[base + 64 + i] = x2 * c + x1 * s;
}

// ---------------------------------------------------------------------------
// Flash attention (causal, GQA 16Q/4KV heads, HD=128), WMMA QK^T and P*V.
// Workgroup = 4 waves, 64-query tile (16 rows/wave); key/value streamed in
// 64-position blocks.  Online softmax; row stats reduced with shfl_xor masks
// 1/2/4/8 which stay inside each 16-lane half (matching the C layout).
// ---------------------------------------------------------------------------
__global__ __launch_bounds__(128) void k_attn(const float* __restrict__ qkv,
                                              unsigned short* __restrict__ out,
                                              int S) {
  const int b = blockIdx.z, h = blockIdx.y;
  const int q0 = blockIdx.x * 64;
  const int kvh = h >> 2;
  const int tid = threadIdx.x;
  const int wave = tid >> 5, lane = tid & 31;
  const int half = lane >> 4, l15 = lane & 15;

  __shared__ unsigned short Qs[64 * 128];
  __shared__ unsigned short Ks[64 * 128];
  __shared__ unsigned short Vt[128 * 64];   // V transposed: [d][kpos]
  __shared__ unsigned short Ps[4][16 * 64]; // per-wave P tile

  // packed pair staging: thread converts two adjacent f32 -> one b32 store
  for (int idx = tid; idx < 64 * 64; idx += 128) {
    int r = idx >> 6, d2 = (idx & 63) * 2;
    const float* src = &qkv[(size_t)(b * S + q0 + r) * 3072 + h * 128 + d2];
    *(unsigned*)&Qs[r * 128 + d2] = pk2bf(src[0], src[1]);
  }
  __syncthreads();

  v16bf aq[4];
#pragma unroll
  for (int ks = 0; ks < 4; ks++)
    aq[ks] = load_frag_a(Qs + wave * 16 * 128 + ks * 32, 128);

  float mrow[8], lrow[8];
  v8f acc[8];
#pragma unroll
  for (int r = 0; r < 8; r++) { mrow[r] = -1e30f; lrow[r] = 0.f; }
#pragma unroll
  for (int dt = 0; dt < 8; dt++) acc[dt] = zero8();

  const float scale = 0.08838834764831845f;  // 1/sqrt(128)
  const int nblocks = q0 / 64 + 1;

  for (int jb = 0; jb < nblocks; jb++) {
    for (int idx = tid; idx < 64 * 64; idx += 128) {
      int r = idx >> 6, d2 = (idx & 63) * 2;
      size_t rowp = (size_t)(b * S + jb * 64 + r) * 3072;
      const float* ksrc = &qkv[rowp + 2048 + kvh * 128 + d2];
      *(unsigned*)&Ks[r * 128 + d2] = pk2bf(ksrc[0], ksrc[1]);
      const float* vsrc = &qkv[rowp + 2560 + kvh * 128 + d2];
      Vt[d2 * 64 + r]       = f2bf(vsrc[0]);
      Vt[(d2 + 1) * 64 + r] = f2bf(vsrc[1]);
    }
    __syncthreads();

    // scores: S_tile[16x64] = Q[16x128] * K_block^T  (K rows serve as BT)
    v8f st[4];
#pragma unroll
    for (int nt = 0; nt < 4; nt++) {
      v8f s = zero8();
#pragma unroll
      for (int ks = 0; ks < 4; ks++) {
        v16bf bk = load_frag_b(Ks + (nt * 16) * 128 + ks * 32, 128);
        s = wmma_bf16(aq[ks], bk, s);
      }
      st[nt] = s;
    }

    float rmax[8];
#pragma unroll
    for (int r = 0; r < 8; r++) rmax[r] = -1e30f;
#pragma unroll
    for (int nt = 0; nt < 4; nt++) {
      int kg = jb * 64 + nt * 16 + l15;
#pragma unroll
      for (int r = 0; r < 8; r++) {
        int qg = q0 + wave * 16 + half * 8 + r;
        float v = st[nt][r] * scale;
        if (kg > qg) v = -1e30f;  // causal mask
        st[nt][r] = v;
        rmax[r] = fmaxf(rmax[r], v);
      }
    }
#pragma unroll
    for (int o = 1; o < 16; o <<= 1)
#pragma unroll
      for (int r = 0; r < 8; r++)
        rmax[r] = fmaxf(rmax[r], __shfl_xor(rmax[r], o, 32));

    float corr[8], rsum[8];
#pragma unroll
    for (int r = 0; r < 8; r++) {
      float mn = fmaxf(mrow[r], rmax[r]);
      corr[r] = __expf(mrow[r] - mn);
      mrow[r] = mn;
      rsum[r] = 0.f;
    }
#pragma unroll
    for (int nt = 0; nt < 4; nt++) {
#pragma unroll
      for (int r = 0; r < 8; r++) {
        float p = __expf(st[nt][r] - mrow[r]);
        rsum[r] += p;
        int m = half * 8 + r;
        Ps[wave][m * 64 + nt * 16 + l15] = f2bf(p);
      }
    }
#pragma unroll
    for (int o = 1; o < 16; o <<= 1)
#pragma unroll
      for (int r = 0; r < 8; r++) rsum[r] += __shfl_xor(rsum[r], o, 32);
#pragma unroll
    for (int r = 0; r < 8; r++) lrow[r] = lrow[r] * corr[r] + rsum[r];
#pragma unroll
    for (int dt = 0; dt < 8; dt++)
#pragma unroll
      for (int r = 0; r < 8; r++) acc[dt][r] *= corr[r];
    __syncthreads();  // Ps visible for DS fragment loads

    // O += P[16x64] * V[64x128]   (Vt serves as BT)
#pragma unroll
    for (int kk = 0; kk < 2; kk++) {
      v16bf ap = load_frag_a(&Ps[wave][0] + kk * 32, 64);
#pragma unroll
      for (int dt = 0; dt < 8; dt++) {
        v16bf bv = load_frag_b(Vt + (dt * 16) * 64 + kk * 32, 64);
        acc[dt] = wmma_bf16(ap, bv, acc[dt]);
      }
    }
    __syncthreads();  // done with Ks/Vt before restage
  }

  // normalize + store bf16 [T, NH*HD]
#pragma unroll
  for (int r = 0; r < 8; r++) {
    float inv = 1.f / lrow[r];
    int m = half * 8 + r;
    size_t rowo = (size_t)(b * S + q0 + wave * 16 + m) * 2048 + h * 128;
#pragma unroll
    for (int dt = 0; dt < 8; dt++)
      out[rowo + dt * 16 + l15] = f2bf(acc[dt][r] * inv);
  }
}

// ---------------------------------------------------------------------------
// Tiny projections (router logits N=8, shared gate N=1): one block per token,
// one wave per output column, lane-strided K with shfl reduction.
// ---------------------------------------------------------------------------
__global__ __launch_bounds__(256) void k_small(const unsigned short* __restrict__ A,
                                               const float* __restrict__ W,
                                               float* __restrict__ C, int N, int K) {
  int t = blockIdx.x;
  int wave = threadIdx.x >> 5, lane = threadIdx.x & 31;
  for (int n = wave; n < N; n += 8) {
    float s = 0.f;
    for (int k = lane; k < K; k += 32)
      s += bf2f(A[(size_t)t * K + k]) * W[(size_t)k * N + n];
#pragma unroll
    for (int o = 16; o > 0; o >>= 1) s += __shfl_xor(s, o, 32);
    if (lane == 0) C[(size_t)t * N + n] = s;
  }
}

// softmax over 8 experts, top-2, renormalize -> dense per-token weights [T,8]
__global__ __launch_bounds__(256) void k_route(const float* __restrict__ logits,
                                               float* __restrict__ w, int T) {
  int t = blockIdx.x * 256 + threadIdx.x;
  if (t >= T) return;
  float l[8], mx = -1e30f;
#pragma unroll
  for (int e = 0; e < 8; e++) { l[e] = logits[t * 8 + e]; mx = fmaxf(mx, l[e]); }
  float sum = 0.f;
#pragma unroll
  for (int e = 0; e < 8; e++) { l[e] = __expf(l[e] - mx); sum += l[e]; }
  int i0 = 0, i1 = -1;
  float v0 = -1.f, v1 = -1.f;
#pragma unroll
  for (int e = 0; e < 8; e++) {
    float p = l[e] / sum;
    if (p > v0) { v1 = v0; i1 = i0; v0 = p; i0 = e; }
    else if (p > v1) { v1 = p; i1 = e; }
  }
  float norm = v0 + v1;
#pragma unroll
  for (int e = 0; e < 8; e++) w[t * 8 + e] = 0.f;
  w[t * 8 + i0] = v0 / norm;
  w[t * 8 + i1] = v1 / norm;
}

// act = silu(a) * b  -> bf16 (packed pair stores; n is even)
__global__ __launch_bounds__(256) void k_silumul(const float* __restrict__ a,
                                                 const float* __restrict__ b,
                                                 unsigned short* __restrict__ o,
                                                 size_t n) {
  for (size_t i = ((size_t)blockIdx.x * 256 + threadIdx.x) * 2; i < n;
       i += (size_t)gridDim.x * 512) {
    float x0 = a[i], x1 = a[i + 1];
    float r0 = (x0 / (1.f + __expf(-x0))) * b[i];
    float r1 = (x1 / (1.f + __expf(-x1))) * b[i + 1];
    *(unsigned*)&o[i] = pk2bf(r0, r1);
  }
}

// ---------------------------------------------------------------------------
extern "C" void kernel_launch(void* const* d_in, const int* in_sizes, int n_in,
                              void* d_out, int out_size, void* d_ws,
                              size_t ws_size, hipStream_t stream) {
  (void)in_sizes; (void)n_in; (void)out_size; (void)ws_size;
  const int B = 2, S = 2048, H = 2048, NH = 16, NKV = 4, HD = 128;
  const int E = 8, F = 1408, FS = 5632;
  const int T = B * S;

  const int*   positions = (const int*)d_in[0];
  const float* hidden    = (const float*)d_in[1];
  const float* ln1       = (const float*)d_in[2];
  const float* ln2       = (const float*)d_in[3];
  const float* wqkv      = (const float*)d_in[4];
  const float* bqkv      = (const float*)d_in[5];
  const float* wo        = (const float*)d_in[6];
  const float* gate_w    = (const float*)d_in[7];
  const float* e_w1      = (const float*)d_in[8];
  const float* e_w3      = (const float*)d_in[9];
  const float* e_w2      = (const float*)d_in[10];
  const float* s_w1      = (const float*)d_in[11];
  const float* s_w3      = (const float*)d_in[12];
  const float* s_w2      = (const float*)d_in[13];
  const float* sg_w      = (const float*)d_in[14];

  float* out    = (float*)d_out;              // [T,H]
  float* resid2 = out + (size_t)T * H;        // [T,H]

  char* ws = (char*)d_ws;
  size_t off = 0;
  auto alloc = [&](size_t bytes) -> void* {
    void* p = ws + off;
    off = (off + bytes + 255) & ~(size_t)255;
    return p;
  };
  unsigned short* xn     = (unsigned short*)alloc((size_t)T * H * 2);
  float*          qkv    = (float*)alloc((size_t)T * 3072 * 4);
  unsigned short* attnbf = (unsigned short*)alloc((size_t)T * H * 2);
  unsigned short* h2     = (unsigned short*)alloc((size_t)T * H * 2);
  float*          big0   = (float*)alloc((size_t)T * FS * 4);
  float*          big1   = (float*)alloc((size_t)T * FS * 4);
  unsigned short* act    = (unsigned short*)alloc((size_t)T * FS * 2);
  float*          logits = (float*)alloc((size_t)T * E * 4);
  float*          routew = (float*)alloc((size_t)T * E * 4);
  float*          sg     = (float*)alloc((size_t)T * 4);

  // 1. RMSNorm1 -> bf16
  k_rms<<<T, 256, 0, stream>>>(hidden, ln1, xn, H);
  // 2. QKV projection (+bias)
  dim3 gq(3072 / 128, T / 128);
  k_gemm<<<gq, 256, 0, stream>>>(xn, wqkv, qkv, T, 3072, H, 1, bqkv, nullptr, 0, 0);
  // 3. RoPE on Q and K heads
  int nrope = T * (NH + NKV) * (HD / 2);
  k_rope<<<(nrope + 255) / 256, 256, 0, stream>>>(qkv, positions, T);
  // 4. Flash attention -> bf16 [T, NH*HD]
  dim3 ga(S / 64, NH, B);
  k_attn<<<ga, 128, 0, stream>>>(qkv, attnbf, S);
  // 5. Output projection + residual -> residual2 (second half of d_out)
  dim3 gh(H / 128, T / 128);
  k_gemm<<<gh, 256, 0, stream>>>(attnbf, wo, resid2, T, H, H, 2, nullptr, hidden, 0, 0);
  // 6. RMSNorm2 -> bf16
  k_rms<<<T, 256, 0, stream>>>(resid2, ln2, h2, H);
  // 7. Router logits, shared-expert gate, top-2 routing weights
  k_small<<<T, 256, 0, stream>>>(h2, gate_w, logits, E, H);
  k_small<<<T, 256, 0, stream>>>(h2, sg_w, sg, 1, H);
  k_route<<<(T + 255) / 256, 256, 0, stream>>>(logits, routew, T);
  // 8. Shared SwiGLU MLP, sigmoid-gated, initializes d_out
  dim3 gs(FS / 128, T / 128);
  k_gemm<<<gs, 256, 0, stream>>>(h2, s_w1, big0, T, FS, H, 0, nullptr, nullptr, 0, 0);
  k_gemm<<<gs, 256, 0, stream>>>(h2, s_w3, big1, T, FS, H, 0, nullptr, nullptr, 0, 0);
  k_silumul<<<4096, 256, 0, stream>>>(big0, big1, act, (size_t)T * FS);
  k_gemm<<<gh, 256, 0, stream>>>(act, s_w2, out, T, H, FS, 3, nullptr, sg, 0, 0);
  // 9. Experts: dense per-expert SwiGLU, accumulate routew[t,e]*(he@w2) into out
  dim3 gf(F / 128, T / 128);
  for (int e = 0; e < E; e++) {
    k_gemm<<<gf, 256, 0, stream>>>(h2, e_w1 + (size_t)e * H * F, big0,
                                   T, F, H, 0, nullptr, nullptr, 0, 0);
    k_gemm<<<gf, 256, 0, stream>>>(h2, e_w3 + (size_t)e * H * F, big1,
                                   T, F, H, 0, nullptr, nullptr, 0, 0);
    k_silumul<<<4096, 256, 0, stream>>>(big0, big1, act, (size_t)T * F);
    k_gemm<<<gh, 256, 0, stream>>>(act, e_w2 + (size_t)e * F * H, out,
                                   T, H, F, 4, nullptr, routew, E, e);
  }
}
